// IPEXQKVFusedGemmInt4OneDNN_47407849013386
// MI455X (gfx1250) — compile-verified, matrix operand order checked by reading, not run on hardware
//
#include <hip/hip_runtime.h>

typedef _Float16 half_t;
typedef __attribute__((ext_vector_type(16))) _Float16 v16h;
typedef __attribute__((ext_vector_type(8)))  _Float16 v8h;
typedef __attribute__((ext_vector_type(8)))  float    v8f;

#define M_TOT 8192
#define K_TOT 4096
#define N_TOT 12288
#define BM 256
#define BN 128
#define BK 64
#define NKT (K_TOT / BK)     // 64 k-tiles
#define LDA 72               // padded LDS stride (halves) -> conflict-free b128 frag reads
#define LDB 72

union ABfrag { v16h v; v8h h[2]; };

__global__ __launch_bounds__(256, 2)
void qkv_int4_wmma(const half_t* __restrict__ X,
                   const int*    __restrict__ QW,
                   const half_t* __restrict__ SC,
                   const int*    __restrict__ QZ,
                   const half_t* __restrict__ BIAS,
                   half_t*       __restrict__ OUT)
{
    // double-buffered tiles: one barrier per K-tile
    __shared__ half_t As[2][BM * LDA];   // 256 x 64 f16 tile of X (padded)
    __shared__ half_t Bs[2][BN * LDB];   // 128 cols x 64 K of dequantized W, column-major

    const int tid  = threadIdx.x;
    const int lane = tid & 31;
    const int wave = tid >> 5;
    const int wm   = wave & 3;        // 4 waves along M (64 rows each)
    const int wn   = wave >> 2;       // 2 waves along N (64 cols each)
    const int M0   = blockIdx.y * BM;
    const int N0   = blockIdx.x * BN;

    v8h    areg[8];                   // staged A: 8 x (8 halves) per thread
    int    qreg[4];                   // staged packed int4 weights per thread
    int    zreg[4];                   // cached zero-points (per quant group)
    half_t sreg[4];                   // cached scales     (per quant group)

    v8f acc[4][4] = {};

    // ---- global load stages (two-stage pipeline) ----
    auto loadA = [&](int k0) {
#pragma unroll
        for (int it = 0; it < 8; ++it) {
            int idx = tid + it * 256;           // 0..2047
            int m   = idx >> 3;                 // 0..255
            int kc  = idx & 7;                  // 8-half chunk within BK
            areg[it] = *(const v8h*)(X + (size_t)(M0 + m) * K_TOT + k0 + kc * 8);
        }
    };
    auto loadB = [&](int k0) {
        const int kp0 = k0 >> 3;                // qweight row base (8 nibbles / int32)
        const bool newGroup = (k0 & 127) == 0;  // blocksize=128: group flips every 2 tiles
        const int g = k0 >> 7;
#pragma unroll
        for (int it = 0; it < 4; ++it) {
            int idx = tid + it * 256;           // 0..1023
            int kpi = idx >> 7;                 // 0..7 (qweight row within tile)
            int c   = idx & 127;                // column within BN
            qreg[it] = QW[(size_t)(kp0 + kpi) * N_TOT + N0 + c];
            if (newGroup) {
                int n = N0 + c;
                zreg[it] = (QZ[(size_t)g * (N_TOT / 8) + (n >> 3)] >> ((n & 7) * 4)) & 15;
                sreg[it] = SC[(size_t)g * N_TOT + n];
            }
        }
    };
    auto storeA = [&](int buf) {
#pragma unroll
        for (int it = 0; it < 8; ++it) {
            int idx = tid + it * 256;
            int m   = idx >> 3;
            int kc  = idx & 7;
            *(v8h*)(&As[buf][m * LDA + kc * 8]) = areg[it];
        }
    };
    // dequant int4 -> f16 and store into B-fragment-friendly (column-major) LDS
    auto storeB = [&](int buf) {
#pragma unroll
        for (int it = 0; it < 4; ++it) {
            int idx = tid + it * 256;
            int kpi = idx >> 7;
            int c   = idx & 127;
            int q   = qreg[it];
            int z   = zreg[it];
            half_t s = sreg[it];
            v8h w;
#pragma unroll
            for (int i = 0; i < 8; ++i) {       // nibble i -> K = kpi*8 + i (GPTQ packing)
                int nib = (q >> (i * 4)) & 15;
                w[i] = (half_t)(nib - z) * s;
            }
            *(v8h*)(&Bs[buf][c * LDB + kpi * 8]) = w;
        }
    };

    // ---- WMMA over one BK=64 LDS tile (two 32-K steps) ----
    auto compute = [&](int buf) {
        const int hi  = lane >> 4;
        const int row = lane & 15;
#pragma unroll
        for (int ks = 0; ks < 2; ++ks) {
            const int kb = ks * 32;
            ABfrag a[4], b[4];
#pragma unroll
            for (int mt = 0; mt < 4; ++mt) {
                // ISA 16-bit A 16x32 layout: regs0-3 = K[hi*8 .. +7], regs4-7 = K[16+hi*8 .. +7]
                const half_t* p = &As[buf][(wm * 64 + mt * 16 + row) * LDA + kb + hi * 8];
                a[mt].h[0] = *(const v8h*)p;
                a[mt].h[1] = *(const v8h*)(p + 16);
            }
#pragma unroll
            for (int nt = 0; nt < 4; ++nt) {
                // B 32x16: lane = N column, same K packing as A
                const half_t* p = &Bs[buf][(wn * 64 + nt * 16 + row) * LDB + kb + hi * 8];
                b[nt].h[0] = *(const v8h*)p;
                b[nt].h[1] = *(const v8h*)(p + 16);
            }
#pragma unroll
            for (int mt = 0; mt < 4; ++mt)
#pragma unroll
                for (int nt = 0; nt < 4; ++nt)
                    acc[mt][nt] = __builtin_amdgcn_wmma_f32_16x16x32_f16(
                        false, a[mt].v, false, b[nt].v,
                        (short)0, acc[mt][nt], false, false);
        }
    };

    // ---- main K loop: ping-pong LDS, one barrier per tile ----
    loadA(0);
    loadB(0);
    for (int kt = 0; kt < NKT; ++kt) {
        const int cur = kt & 1;
        storeA(cur);                  // write tile kt (read at kt-2 was fenced by kt-1 barrier)
        storeB(cur);
        __syncthreads();              // buf[cur] visible to all waves
        if (kt + 1 < NKT) {           // overlap next tile's global loads with the WMMA burst
            loadA((kt + 1) * BK);
            loadB((kt + 1) * BK);
        }
        compute(cur);
    }

    // ---- epilogue: +bias, f32->f16, scatter into q/k/v regions ----
    const int hi  = lane >> 4;
    const int row = lane & 15;
#pragma unroll
    for (int nt = 0; nt < 4; ++nt) {
        int gn = N0 + wn * 64 + nt * 16 + row;
        float bv = (float)BIAS[gn];
        int region = gn >> 12;                  // 0=q, 1=k, 2=v (BN=128 divides 4096)
        int col    = gn & 4095;
        size_t base = (size_t)region * ((size_t)M_TOT * 4096) + col;
#pragma unroll
        for (int mt = 0; mt < 4; ++mt) {
            int gm0 = M0 + wm * 64 + mt * 16 + hi * 8;  // C/D layout: VGPR r -> M = hi*8 + r
#pragma unroll
            for (int r = 0; r < 8; ++r) {
                float v = acc[mt][nt][r] + bv;
                OUT[base + (size_t)(gm0 + r) * 4096] = (half_t)v;
            }
        }
    }
}

extern "C" void kernel_launch(void* const* d_in, const int* in_sizes, int n_in,
                              void* d_out, int out_size, void* d_ws, size_t ws_size,
                              hipStream_t stream) {
    const half_t* X    = (const half_t*)d_in[0];
    const int*    QW   = (const int*)   d_in[1];
    const half_t* SC   = (const half_t*)d_in[2];
    const int*    QZ   = (const int*)   d_in[3];
    const half_t* BIAS = (const half_t*)d_in[4];
    // d_in[5] = blocksize (128), compiled in.
    half_t* OUT = (half_t*)d_out;

    dim3 grid(N_TOT / BN, M_TOT / BM);          // 96 x 32 workgroups
    qkv_int4_wmma<<<grid, 256, 0, stream>>>(X, QW, SC, QZ, BIAS, OUT);
}
